// EquivSetGNN3_63153199120612
// MI455X (gfx1250) — matrix operand order repaired and checked by reading, not compiled, not run on hardware
//
#include <hip/hip_runtime.h>
#include <hip/hip_bf16.h>

// ---------------------------------------------------------------------------
// EquivSetGNN on MI455X (gfx1250), bf16 WMMA + scatter-add formulation.
//
//   x  = relu(x @ W_in + b_in);  x0 = x
//   repeat 2x:
//     y  = x @ W1 + b1
//     Xe[e] += y[v]      for (v,e) in incidence        (scatter-add)
//     S [v] += Xe[e]     for (v,e) in incidence        (scatter-add)
//     t  = 0.5*( (deg⊙x) @ W2a + S @ W2b + deg⊙b2 ) + 0.5*x0
//     x  = relu(t @ W3 + b3)
//
// The per-nnz concat GEMM of the reference is algebraically folded into two
// dense N x 128 GEMMs (linearity of gather/segment_sum), leaving the sparse
// phase as pure f32 scatter-adds (HBM/atomic bound) and the dense phase on
// the WMMA matrix pipes.
// ---------------------------------------------------------------------------

typedef __attribute__((ext_vector_type(16))) __bf16 v16bf;
typedef __attribute__((ext_vector_type(8)))  __bf16 v8bf;
typedef __attribute__((ext_vector_type(8)))  float  v8f;

#define GNN_C 128

// ------------------------------- utility kernels ---------------------------

__global__ void gnn_zero_kernel(float4* __restrict__ p, int n4) {
    for (int i = blockIdx.x * blockDim.x + threadIdx.x; i < n4;
         i += gridDim.x * blockDim.x)
        p[i] = make_float4(0.f, 0.f, 0.f, 0.f);
}

__global__ void gnn_deg_kernel(const int* __restrict__ V,
                               float* __restrict__ deg, int nnz) {
    int i = blockIdx.x * blockDim.x + threadIdx.x;
    if (i < nnz) atomicAdd(&deg[V[i]], 1.0f);
}

// x[m, :] *= deg[m]
__global__ void gnn_rowscale_kernel(float* __restrict__ x,
                                    const float* __restrict__ deg, int n) {
    int i = blockIdx.x * blockDim.x + threadIdx.x;
    if (i < n) x[i] *= deg[i >> 7];   // C = 128
}

// dst[sidx[i], :] += src[gidx[i], :]   (one thread per (i, 4-col chunk))
__global__ void gnn_scatter_add_kernel(const float* __restrict__ src,
                                       const int* __restrict__ gidx,
                                       const int* __restrict__ sidx,
                                       float* __restrict__ dst, int nnz) {
    int tid = blockIdx.x * blockDim.x + threadIdx.x;
    int i = tid >> 5;            // incidence index
    int c = (tid & 31) << 2;     // column (x4)
    if (i >= nnz) return;
    int g = gidx[i];
    int s = sidx[i];
    const float4 v = *(const float4*)(src + (size_t)g * GNN_C + c);
    float* d = dst + (size_t)s * GNN_C + c;
    atomicAdd(d + 0, v.x);
    atomicAdd(d + 1, v.y);
    atomicAdd(d + 2, v.z);
    atomicAdd(d + 3, v.w);
}

// ------------------------------- WMMA GEMM ---------------------------------
//
// out[m, n] = epilogue( sum_k A(m,k) * W(k,n) )   for m < nrows, n < 128.
//   A(m,k) = A0[m,k] (k < 128)  |  A1[m,k-128] (k >= 128, K==256 concat GEMM)
// epilogue:
//   MIX == false :  v = acc + bias[n]                     (+ optional relu)
//   MIX == true  :  v = 0.5*(acc + rowscale[m]*bias[n]) + 0.5*mix[m,n]
//
// Block: 256 threads = 8 waves; block covers 128 rows, each wave a 16-row
// strip x all 128 cols (8 accumulators of 16x16).  W is staged in LDS as
// bf16, transposed [n][k] with +8 pad (conflict-free ds_load_b128 reads).
// K templated -> fully unrolled k-loop; A-fragment global loads are software
// pipelined one k-step ahead so their latency hides behind the 8-wmma burst.
// ---------------------------------------------------------------------------

template <int K, bool MIX, bool RELU>
__global__ __launch_bounds__(256)
void gnn_gemm_kernel(const float* __restrict__ A0, const float* __restrict__ A1,
                     const float* __restrict__ W,  const float* __restrict__ bias,
                     const float* __restrict__ rowscale,
                     const float* __restrict__ mix,
                     float* __restrict__ out, int nrows) {
    extern __shared__ __bf16 ldsW[];            // [128][K+8]
    const int tid = threadIdx.x;
    constexpr int STRIDE = K + 8;

    // Stage W^T into LDS as bf16: ldsW[n*STRIDE + k] = W[k*128 + n]
#pragma unroll 4
    for (int idx = tid; idx < K * GNN_C; idx += 256) {
        int k = idx >> 7;
        int n = idx & 127;
        ldsW[n * STRIDE + k] = (__bf16)W[idx];
    }
    __syncthreads();

    const int wave   = tid >> 5;
    const int lane   = tid & 31;
    const int laneLo = lane & 15;
    const int laneHi = lane >> 4;
    const int mbase  = blockIdx.x * 128 + wave * 16;
    if (mbase >= nrows) return;                 // uniform per wave

    const int rowA = min(mbase + laneLo, nrows - 1);
    const size_t rowOff = (size_t)rowA * GNN_C + laneHi * 8;

    if (MIX) {
        // stream the restart features toward L2 early (global_prefetch_b8)
        __builtin_prefetch(mix + (size_t)rowA * GNN_C, 0, 1);
    }

    v8f acc[8];
#pragma unroll
    for (int t = 0; t < 8; ++t) acc[t] = (v8f){0, 0, 0, 0, 0, 0, 0, 0};

    // ---- prologue: issue A-fragment loads for k-step 0
    float4 f0, f1, f2, f3;
    {
        const float* Ar = A0 + rowOff;
        f0 = ((const float4*)Ar)[0];
        f1 = ((const float4*)Ar)[1];
        f2 = ((const float4*)(Ar + 16))[0];
        f3 = ((const float4*)(Ar + 16))[1];
    }

#pragma unroll
    for (int kb = 0; kb < K; kb += 32) {
        // ---- pack current A fragment (16x32 bf16): lane holds row m=laneLo,
        //      k = (e&7) + (e>>3)*16 + laneHi*8 + kb  -> two contiguous 8-runs
        v16bf a;
        a[0]  = (__bf16)f0.x; a[1]  = (__bf16)f0.y;
        a[2]  = (__bf16)f0.z; a[3]  = (__bf16)f0.w;
        a[4]  = (__bf16)f1.x; a[5]  = (__bf16)f1.y;
        a[6]  = (__bf16)f1.z; a[7]  = (__bf16)f1.w;
        a[8]  = (__bf16)f2.x; a[9]  = (__bf16)f2.y;
        a[10] = (__bf16)f2.z; a[11] = (__bf16)f2.w;
        a[12] = (__bf16)f3.x; a[13] = (__bf16)f3.y;
        a[14] = (__bf16)f3.z; a[15] = (__bf16)f3.w;

        // ---- software pipeline: issue next k-step's A loads NOW, so their
        //      latency hides behind this step's ds loads + wmma burst.
        if (kb + 32 < K) {
            const int kn = kb + 32;
            const float* Abase = (kn < 128) ? A0 : A1;   // folds after unroll
            const float* Ar = Abase + rowOff + (kn & 127);
            f0 = ((const float4*)Ar)[0];
            f1 = ((const float4*)Ar)[1];
            f2 = ((const float4*)(Ar + 16))[0];
            f3 = ((const float4*)(Ar + 16))[1];
        }

        // ---- all 8 B fragments (batched ds_load_b128), then 8-wmma burst.
        //      B (32x16 bf16): lane holds col n=laneLo, k = kb + (e&7) + (e>>3)*16 + laneHi*8
        const __bf16* bp = ldsW + kb + laneHi * 8;
        v16bf bfr[8];
#pragma unroll
        for (int t = 0; t < 8; ++t) {
            const __bf16* p = bp + (t * 16 + laneLo) * STRIDE;
            v8bf lo = *(const v8bf*)p;
            v8bf hi = *(const v8bf*)(p + 16);
#pragma unroll
            for (int e = 0; e < 8; ++e) { bfr[t][e] = lo[e]; bfr[t][e + 8] = hi[e]; }
        }
#pragma unroll
        for (int t = 0; t < 8; ++t) {
            acc[t] = __builtin_amdgcn_wmma_f32_16x16x32_bf16(
                false, a, false, bfr[t], (short)0, acc[t], false, false);
        }
    }

    // ---- epilogue: C/D layout -> VGPR r: m = r + 8*laneHi, n = 16t + laneLo
#pragma unroll
    for (int t = 0; t < 8; ++t) {
        const int n = t * 16 + laneLo;
        const float bv = bias[n];
#pragma unroll
        for (int r = 0; r < 8; ++r) {
            const int m = mbase + r + laneHi * 8;
            if (m < nrows) {
                float v = acc[t][r];
                if (MIX) {
                    v = 0.5f * (v + rowscale[m] * bv)
                      + 0.5f * mix[(size_t)m * GNN_C + n];
                } else {
                    v += bv;
                }
                if (RELU) v = fmaxf(v, 0.0f);
                out[(size_t)m * GNN_C + n] = v;
            }
        }
    }
}

// ------------------------------- launcher ----------------------------------

extern "C" void kernel_launch(void* const* d_in, const int* in_sizes, int n_in,
                              void* d_out, int out_size, void* d_ws, size_t ws_size,
                              hipStream_t stream) {
    const float* x_in = (const float*)d_in[0];
    const float* W_in = (const float*)d_in[1];
    const float* b_in = (const float*)d_in[2];
    const float* W1   = (const float*)d_in[3];
    const float* b1   = (const float*)d_in[4];
    const float* W2   = (const float*)d_in[5];   // [256,128]: rows 0..127 -> x, 128..255 -> Xe
    const float* b2   = (const float*)d_in[6];
    const float* W3   = (const float*)d_in[7];
    const float* b3   = (const float*)d_in[8];
    const int*   V    = (const int*)d_in[9];
    const int*   E    = (const int*)d_in[10];

    const int N   = in_sizes[0] / GNN_C;   // 50000
    const int NNZ = in_sizes[9];           // 800000
    const int M   = 25000;                 // n_edges (reference constant)

    // workspace layout (floats)
    float* A   = (float*)d_ws;             // N*C   current features / xd / t
    float* B   = A  + (size_t)N * GNN_C;   // N*C   y / S / next features
    float* x0  = B  + (size_t)N * GNN_C;   // N*C   restart features
    float* Xe  = x0 + (size_t)N * GNN_C;   // M*C   hyperedge features
    float* deg = Xe + (size_t)M * GNN_C;   // N     node degrees

    const int gemm_grid = (N + 127) / 128;
    const size_t sh128 = (size_t)GNN_C * (128 + 8) * sizeof(__bf16);
    const size_t sh256 = (size_t)GNN_C * (256 + 8) * sizeof(__bf16);
    const int scat_grid = (NNZ * 32 + 255) / 256;

    // deg[v] = multiplicity of v in V   (recomputed every call: replay-safe)
    gnn_zero_kernel<<<64, 256, 0, stream>>>((float4*)deg, N / 4);
    gnn_deg_kernel<<<(NNZ + 255) / 256, 256, 0, stream>>>(V, deg, NNZ);

    // x = relu(x_in @ W_in + b_in);  x0 = x
    gnn_gemm_kernel<128, false, true><<<gemm_grid, 256, sh128, stream>>>(
        x_in, nullptr, W_in, b_in, nullptr, nullptr, A, N);
    hipMemcpyAsync(x0, A, (size_t)N * GNN_C * sizeof(float),
                   hipMemcpyDeviceToDevice, stream);

    float* cur = A;
    float* oth = B;
    for (int layer = 0; layer < 2; ++layer) {
        // y = x @ W1 + b1
        gnn_gemm_kernel<128, false, false><<<gemm_grid, 256, sh128, stream>>>(
            cur, nullptr, W1, b1, nullptr, nullptr, oth, N);
        // Xe = segment_sum(y[V], E)
        gnn_zero_kernel<<<2048, 256, 0, stream>>>((float4*)Xe, M * GNN_C / 4);
        gnn_scatter_add_kernel<<<scat_grid, 256, 0, stream>>>(oth, V, E, Xe, NNZ);
        // S = segment_sum(Xe[E], V)   (reuses `oth`)
        gnn_zero_kernel<<<2048, 256, 0, stream>>>((float4*)oth, N * GNN_C / 4);
        gnn_scatter_add_kernel<<<scat_grid, 256, 0, stream>>>(Xe, E, V, oth, NNZ);
        // xd = deg ⊙ x   (in place on cur)
        gnn_rowscale_kernel<<<(N * GNN_C + 255) / 256, 256, 0, stream>>>(cur, deg, N * GNN_C);
        // t = 0.5*(concat(xd,S) @ W2 + deg⊙b2) + 0.5*x0   (in place on cur;
        // safe: each wave reads/writes exactly its own 16-row strip)
        gnn_gemm_kernel<256, true, false><<<gemm_grid, 256, sh256, stream>>>(
            cur, oth, W2, b2, deg, x0, cur, N);
        // x = relu(t @ W3 + b3)
        gnn_gemm_kernel<128, false, true><<<gemm_grid, 256, sh128, stream>>>(
            cur, nullptr, W3, b3, nullptr, nullptr, oth, N);
        float* tmp = cur; cur = oth; oth = tmp;
    }

    hipMemcpyAsync(d_out, cur, (size_t)N * GNN_C * sizeof(float),
                   hipMemcpyDeviceToDevice, stream);
}
